// RotateLinear_88897233092672
// MI455X (gfx1250) — matrix operand order repaired and checked by reading, not compiled, not run on hardware
//
#include <hip/hip_runtime.h>
#include <hip/hip_bf16.h>
#include <stdint.h>

// out = FWHT(x)/64 @ W^T + b   (H is Sylvester Hadamard / sqrt(4096), symmetric)
// (1) rowwise FWHT in LDS, scale, cast f16 -> ws
// (2) weight f32 -> f16 -> ws
// (3) f16 WMMA GEMM (M=8192,N=4096,K=4096): triple-buffered LDS pipeline fed by
//     GLOBAL_LOAD_ASYNC_TO_LDS_B128 (ASYNCcnt), one barrier per 16 WMMAs,
//     fp32 accum, fused bias.

typedef _Float16 f16;
typedef __attribute__((ext_vector_type(16))) _Float16 v16h;
typedef __attribute__((ext_vector_type(8)))  float    v8f;
typedef int v4i __attribute__((vector_size(16)));

#define D_MODEL 4096
#define D_OUT   4096
#define M_TOT   (4 * 2048)      // BATCH * SEQ
#define KT      64              // K per LDS tile (two WMMA k-steps)
#define LDA     72              // padded LDS row stride in f16 (36 banks -> conflict-free)
#define TILE_ELEMS (128 * LDA)  // f16 elems per matrix per buffer
#define SMEM_BYTES (6 * TILE_ELEMS * 2)  // 3 buffers x {A,B} = 108 KB (dynamic)

#if defined(__has_builtin)
#  if __has_builtin(__builtin_amdgcn_global_load_async_to_lds_b128)
#    define HAS_ASYNC_LDS 1
#  endif
#  if __has_builtin(__builtin_amdgcn_s_wait_asynccnt)
#    define HAS_WAIT_BUILTIN 1
#  endif
#endif

#if defined(HAS_ASYNC_LDS)
#  if defined(HAS_WAIT_BUILTIN)
#    define WAIT_ASYNC(n) __builtin_amdgcn_s_wait_asynccnt(n)
#  else
#    define WAIT_ASYNC(n) asm volatile("s_wait_asynccnt %0" ::"i"(n) : "memory")
#  endif
#else
#  define WAIT_ASYNC(n) ((void)0)
#endif

// 16-byte global -> LDS copy: GLOBAL_LOAD_ASYNC_TO_LDS_B128, tracked by ASYNCcnt.
// Builtin signature (verified by hipcc diagnostic): (int4 AS(1)*, int4 AS(3)*, Ii, Ii).
__device__ __forceinline__ void cp16_g2l(const f16* g, f16* l) {
#if defined(HAS_ASYNC_LDS)
  typedef __attribute__((address_space(1))) v4i GInt4;
  typedef __attribute__((address_space(3))) v4i LInt4;
  __builtin_amdgcn_global_load_async_to_lds_b128(
      (GInt4*)(uintptr_t)g, (LInt4*)(uint32_t)(uintptr_t)l, 0, 0);
#else
  *(uint4*)l = *(const uint4*)g;
#endif
}

union FragU { uint4 u[2]; v16h v; };

// ---------------- Kernel 1: rowwise FWHT + scale + cast to f16 ----------------
__global__ __launch_bounds__(256) void fwht_f16_kernel(const float* __restrict__ x,
                                                       f16* __restrict__ xr) {
  __shared__ float row[D_MODEL];
  const size_t m = blockIdx.x;
  const float* src = x + m * (size_t)D_MODEL;
  for (int i = threadIdx.x; i < D_MODEL; i += 256) row[i] = src[i];
  __syncthreads();
#pragma unroll 1
  for (int s = 0; s < 12; ++s) {
    const int h = 1 << s;
    for (int t = threadIdx.x; t < D_MODEL / 2; t += 256) {
      const int i = ((t >> s) << (s + 1)) | (t & (h - 1));
      const int j = i | h;
      const float a = row[i];
      const float b = row[j];
      row[i] = a + b;
      row[j] = a - b;
    }
    __syncthreads();
  }
  f16* dst = xr + m * (size_t)D_MODEL;
  const float scale = 0.015625f;  // 1/sqrt(4096)
  for (int i = threadIdx.x; i < D_MODEL; i += 256)
    dst[i] = (f16)(row[i] * scale);
}

// ---------------- Kernel 2: weight f32 -> f16 ----------------
__global__ __launch_bounds__(256) void wcvt_kernel(const float* __restrict__ w,
                                                   f16* __restrict__ wb) {
  const size_t i = ((size_t)blockIdx.x * 256 + threadIdx.x) * 4;
  const float4 v = *(const float4*)(w + i);
  union { f16 h[4]; uint2 u; } pk;
  pk.h[0] = (f16)v.x; pk.h[1] = (f16)v.y; pk.h[2] = (f16)v.z; pk.h[3] = (f16)v.w;
  *(uint2*)(wb + i) = pk.u;
}

// ---------------- Kernel 3: f16 WMMA GEMM, triple-buffered LDS ----------------
// A: xr [M, K] row-major f16.  B: wb [N, K] row-major f16 (== weight layout).
// Block tile 128x128, 8 waves (2x4), wave tile 64x32 = 4x2 fragments of 16x16.
__global__ __launch_bounds__(256) void gemm_wmma_kernel(const f16* __restrict__ A,
                                                        const f16* __restrict__ B,
                                                        const float* __restrict__ bias,
                                                        float* __restrict__ C) {
  extern __shared__ __align__(16) f16 smem[];  // 3x{A,B} tile buffers

  const int tid  = threadIdx.x;
  const int lane = tid & 31;
  const int wave = tid >> 5;
  const int hi   = lane >> 4;    // lane half (0/1)
  const int r    = lane & 15;
  const int wm   = wave >> 2;    // 0..1
  const int wn   = wave & 3;     // 0..3
  const int mBase = blockIdx.y * 128;
  const int nBase = blockIdx.x * 128;
  const size_t K = D_MODEL;

  // Staging: 1024 16B chunks per matrix per K-tile; 256 threads -> 4 each.
  // Chunk c = tid + k*256: row = c>>3 (0..127), col = (c&7)*8 f16.
  const f16* gA[4];
  const f16* gB[4];
  int lofs[4];
#pragma unroll
  for (int k = 0; k < 4; ++k) {
    const int c = tid + k * 256;
    const int srow = c >> 3;
    const int scol = (c & 7) * 8;
    gA[k] = A + (size_t)(mBase + srow) * K + scol;
    gB[k] = B + (size_t)(nBase + srow) * K + scol;
    lofs[k] = srow * LDA + scol;
  }

  f16* sAb[3];
  f16* sBb[3];
#pragma unroll
  for (int b = 0; b < 3; ++b) {
    sAb[b] = smem + b * TILE_ELEMS;
    sBb[b] = smem + (3 + b) * TILE_ELEMS;
  }

  v8f acc[4][2];
  const v8f zero = {};
#pragma unroll
  for (int i = 0; i < 4; ++i)
#pragma unroll
    for (int j = 0; j < 2; ++j) acc[i][j] = zero;

  float bv[2];
  bv[0] = bias[nBase + wn * 32 + r];
  bv[1] = bias[nBase + wn * 32 + 16 + r];

  // LDS fragment offsets (ISA 7.12.2 16-bit A layout; B == A-layout on B^T):
  // A frag i: row wm*64+i*16+r; 8-f16 chunks at ks+hi*8 and ks+16+hi*8.
  // B frag j: row wn*32+j*16+r; 8-f16 chunks at ks+hi*8 and ks+16+hi*8.
  int aoff[4], boff[2];
#pragma unroll
  for (int i = 0; i < 4; ++i) aoff[i] = (wm * 64 + i * 16 + r) * LDA + hi * 8;
#pragma unroll
  for (int j = 0; j < 2; ++j) boff[j] = (wn * 32 + j * 16 + r) * LDA + hi * 8;

  // --- helpers (inlined) ---
  auto stage = [&](int kt, f16* dA, f16* dB) {
#pragma unroll
    for (int k = 0; k < 4; ++k) {
      cp16_g2l(gA[k] + kt, dA + lofs[k]);
      cp16_g2l(gB[k] + kt, dB + lofs[k]);
    }
  };
  auto consume = [&](const f16* cA, const f16* cB) {
#pragma unroll
    for (int ks = 0; ks < KT; ks += 32) {
      FragU a[4], b[2];
#pragma unroll
      for (int i = 0; i < 4; ++i) {
        a[i].u[0] = *(const uint4*)&cA[aoff[i] + ks];
        a[i].u[1] = *(const uint4*)&cA[aoff[i] + ks + 16];
      }
#pragma unroll
      for (int j = 0; j < 2; ++j) {
        b[j].u[0] = *(const uint4*)&cB[boff[j] + ks];
        b[j].u[1] = *(const uint4*)&cB[boff[j] + ks + 16];
      }
#pragma unroll
      for (int i = 0; i < 4; ++i)
#pragma unroll
        for (int j = 0; j < 2; ++j)
          acc[i][j] = __builtin_amdgcn_wmma_f32_16x16x32_f16(
              false, a[i].v, false, b[j].v, (short)0, acc[i][j], false, false);
    }
  };

  // Prologue: tiles 0 and 1 in flight.
  stage(0, sAb[0], sBb[0]);
  stage(KT, sAb[1], sBb[1]);

  f16 *aCur = sAb[0], *aNxt = sAb[1], *aFree = sAb[2];
  f16 *bCur = sBb[0], *bNxt = sBb[1], *bFree = sBb[2];

  const int T = D_MODEL / KT;  // 64 K-tiles
#pragma unroll 1
  for (int t = 0; t < T - 2; ++t) {
    WAIT_ASYNC(8);     // tile t landed (tile t+1's 8 copies may remain in flight)
    __syncthreads();   // all waves: tile t in LDS; all reads of tile t-1 done
    stage((t + 2) * KT, aFree, bFree);  // overwrites tile t-1's buffer: safe
    consume(aCur, bCur);
    f16* ta = aCur; aCur = aNxt; aNxt = aFree; aFree = ta;
    f16* tb = bCur; bCur = bNxt; bNxt = bFree; bFree = tb;
  }
  // Tail: tiles T-2 and T-1 (no more staging).
  WAIT_ASYNC(8);
  __syncthreads();
  consume(aCur, bCur);
  WAIT_ASYNC(0);
  __syncthreads();
  consume(aNxt, bNxt);

  // C layout: VGPR u, lanes 0-15 -> M=u, lanes 16-31 -> M=8+u; N=lane%16.
  const int m0 = mBase + wm * 64;
  const int n0 = nBase + wn * 32;
#pragma unroll
  for (int i = 0; i < 4; ++i) {
#pragma unroll
    for (int j = 0; j < 2; ++j) {
      const int col = n0 + j * 16 + r;
      float* cp = C + (size_t)(m0 + i * 16 + hi * 8) * D_OUT + col;
#pragma unroll
      for (int u = 0; u < 8; ++u)
        __builtin_nontemporal_store(acc[i][j][u] + bv[j], cp + (size_t)u * D_OUT);
    }
  }
}

extern "C" void kernel_launch(void* const* d_in, const int* in_sizes, int n_in,
                              void* d_out, int out_size, void* d_ws, size_t ws_size,
                              hipStream_t stream) {
  const float* x      = (const float*)d_in[0];  // [4,2048,4096]
  const float* weight = (const float*)d_in[1];  // [4096,4096]
  const float* bias   = (const float*)d_in[2];  // [4096]
  // d_in[3] = had_K: unused — replaced by exact FWHT.
  float* out = (float*)d_out;

  f16* xr = (f16*)d_ws;                    // 8192*4096 f16 = 64 MB
  f16* wb = xr + (size_t)M_TOT * D_MODEL;  // 4096*4096 f16 = 32 MB

  fwht_f16_kernel<<<M_TOT, 256, 0, stream>>>(x, xr);
  wcvt_kernel<<<(int)(((size_t)D_OUT * D_MODEL / 4) / 256), 256, 0, stream>>>(weight, wb);
  gemm_wmma_kernel<<<dim3(D_OUT / 128, M_TOT / 128), 256, SMEM_BYTES, stream>>>(
      xr, wb, bias, out);
}
